// TripletAttention_21809843929827
// MI455X (gfx1250) — matrix operand order repaired and checked by reading, not compiled, not run on hardware
//
#include <hip/hip_runtime.h>
#include <hip/hip_bf16.h>

typedef __attribute__((ext_vector_type(16))) _Float16 v16h;
typedef __attribute__((ext_vector_type(8)))  _Float16 v8h;
typedef __attribute__((ext_vector_type(8)))  float    v8f;

#define WMMA_F16(a, b, c) \
  __builtin_amdgcn_wmma_f32_16x16x32_f16(false, (a), false, (b), (short)0, (c), false, false)

// ---------------------------------------------------------------------------
// WMMA operand loaders (CDNA5 16-bit layouts, CDNA5 ISA §7.12.2)
// Both are two contiguous 16-byte loads per lane -> global_load_b128.
// ---------------------------------------------------------------------------

// A operand: 16x32 (MxK) tile from row-major f16 src (leading dim ld, elements).
// lanes 0-15: M=lane, K chunks {8*half..+7} in slots 0..7, {16+8*half..+7} in 8..15.
__device__ __forceinline__ v16h load_a16(const _Float16* __restrict__ src, int ld,
                                         int row0, int k0, int lane) {
  int m = lane & 15, half = lane >> 4;
  const _Float16* p = src + (size_t)(row0 + m) * ld + k0 + 8 * half;
  v8h lo = *(const v8h*)(p);
  v8h hi = *(const v8h*)(p + 16);
  v16h r;
#pragma unroll
  for (int e = 0; e < 8; ++e) { r[e] = lo[e]; r[8 + e] = hi[e]; }
  return r;
}

// B operand (32x16 KxN) for C = A * B^T: operand(k,n) = src[n0+n][k0+k]
// (src row-major, ld). lanes 0-15: N=lane, K chunks {8*half..+7} in slots 0..7, {16+8*half..+7} in 8..15.
__device__ __forceinline__ v16h load_bt16(const _Float16* __restrict__ src, int ld,
                                          int n0, int k0, int lane) {
  int n = lane & 15, half = lane >> 4;
  const _Float16* p = src + (size_t)(n0 + n) * ld + k0 + 8 * half;
  v8h lo = *(const v8h*)(p);
  v8h hi = *(const v8h*)(p + 16);
  v16h r;
#pragma unroll
  for (int e = 0; e < 8; ++e) { r[e] = lo[e]; r[8 + e] = hi[e]; }
  return r;
}

// D/C layout: element acc[v] lives at (m = v + 8*(lane>>4), n = lane&15).

// ---------------------------------------------------------------------------
// Kernels
// ---------------------------------------------------------------------------

__global__ void k_f32_to_f16(const float* __restrict__ s, _Float16* __restrict__ d, int n) {
  int i = blockIdx.x * blockDim.x + threadIdx.x;
  if (i < n) d[i] = (_Float16)s[i];
}

// dst[n*rows + k] = (f16) src[k*cols + n]   (transpose convert, coalesced read)
__global__ void k_f32_to_f16_T(const float* __restrict__ s, _Float16* __restrict__ d,
                               int rows, int cols) {
  int i = blockIdx.x * blockDim.x + threadIdx.x;
  if (i >= rows * cols) return;
  int k = i / cols, n = i % cols;
  d[(size_t)n * rows + k] = (_Float16)s[i];
}

// qkv: w = xh(256x128) @ WhT(512x128)^T; scatter f16 into abc[q][h][256][32],
// v stored transposed: vT[h][32][256] (region q==3 of same buffer)
__global__ void k_qkv(const _Float16* __restrict__ xh, const _Float16* __restrict__ whT,
                      _Float16* __restrict__ abcv) {
  int lane = threadIdx.x & 31;
  int wave = blockIdx.x * 4 + (threadIdx.x >> 5);  // 0..511
  int tm = wave >> 5, tn = wave & 31;
  v8f acc = {};
#pragma unroll
  for (int kk = 0; kk < 4; ++kk) {
    v16h A = load_a16(xh, 128, tm * 16, kk * 32, lane);
    v16h B = load_bt16(whT, 128, tn * 16, kk * 32, lane);
    acc = WMMA_F16(A, B, acc);
  }
  int nl = lane & 15, half = lane >> 4;
#pragma unroll
  for (int vv = 0; vv < 8; ++vv) {
    int m = tm * 16 + vv + 8 * half;
    int co = tn * 16 + nl;
    int q = co >> 7, h = (co >> 5) & 3, dd = co & 31;
    if (q < 3)
      abcv[(size_t)(q * 4 + h) * 8192 + m * 32 + dd] = (_Float16)acc[vv];
    else  // v, transposed: vT[h][dd][m]
      abcv[(size_t)(12 + h) * 8192 + dd * 256 + m] = (_Float16)acc[vv];
  }
}

// P = exp(s a.b^T), Q = exp(s a.c^T), R = exp(s b.c^T), Rt = exp(s c.b^T) = R^T
__global__ void k_scores(const _Float16* __restrict__ abcv, _Float16* __restrict__ P,
                         _Float16* __restrict__ Q, _Float16* __restrict__ R,
                         _Float16* __restrict__ Rt) {
  int lane = threadIdx.x & 31;
  int wave = blockIdx.x * 4 + (threadIdx.x >> 5);  // 0..4095
  int tile = wave & 255;
  int sel = wave >> 8;             // 0..15
  int head = sel & 3, mat = sel >> 2;
  int tm = tile >> 4, tn = tile & 15;
  const _Float16* a = abcv + (size_t)(0 + head) * 8192;
  const _Float16* b = abcv + (size_t)(4 + head) * 8192;
  const _Float16* c = abcv + (size_t)(8 + head) * 8192;
  const _Float16* X; const _Float16* Y; _Float16* D;
  if (mat == 0)      { X = a; Y = b; D = P;  }
  else if (mat == 1) { X = a; Y = c; D = Q;  }
  else if (mat == 2) { X = b; Y = c; D = R;  }
  else               { X = c; Y = b; D = Rt; }
  v16h A = load_a16(X, 32, tm * 16, 0, lane);
  v16h B = load_bt16(Y, 32, tn * 16, 0, lane);
  v8f acc = {};
  acc = WMMA_F16(A, B, acc);
  const float scale = 0.17677669529663687f;  // 1/sqrt(32)
  int nl = lane & 15, half = lane >> 4;
  _Float16* dst = D + (size_t)head * 65536;
#pragma unroll
  for (int vv = 0; vv < 8; ++vv) {
    int m = tm * 16 + vv + 8 * half, n = tn * 16 + nl;
    dst[m * 256 + n] = (_Float16)__expf(acc[vv] * scale);
  }
}

// U = P @ R = P @ (Rt)^T  (per head, K=256), f32 out
__global__ void k_U(const _Float16* __restrict__ P, const _Float16* __restrict__ Rt,
                    float* __restrict__ U) {
  int lane = threadIdx.x & 31;
  int wave = blockIdx.x * 4 + (threadIdx.x >> 5);  // 0..1023
  int head = wave >> 8, tile = wave & 255;
  int tm = tile >> 4, tn = tile & 15;
  const _Float16* Ph = P + (size_t)head * 65536;
  const _Float16* Rh = Rt + (size_t)head * 65536;
  v8f acc = {};
#pragma unroll
  for (int kk = 0; kk < 8; ++kk) {
    if (kk < 7) {
      __builtin_prefetch(Ph + (size_t)(tm * 16 + (lane & 15)) * 256 + (kk + 1) * 32, 0, 1);
      __builtin_prefetch(Rh + (size_t)(tn * 16 + (lane & 15)) * 256 + (kk + 1) * 32, 0, 1);
    }
    v16h A = load_a16(Ph, 256, tm * 16, kk * 32, lane);
    v16h B = load_bt16(Rh, 256, tn * 16, kk * 32, lane);
    acc = WMMA_F16(A, B, acc);
  }
  float* Uh = U + (size_t)head * 65536;
  int nl = lane & 15, half = lane >> 4;
#pragma unroll
  for (int vv = 0; vv < 8; ++vv)
    Uh[(tm * 16 + vv + 8 * half) * 256 + tn * 16 + nl] = acc[vv];
}

// norminv[h][k] = 1 / max(sum_i Q_ik U_ik, 1e-6)
__global__ void k_norminv(const _Float16* __restrict__ Q, const float* __restrict__ U,
                          float* __restrict__ ninv) {
  int head = blockIdx.x, k = threadIdx.x;
  const _Float16* Qh = Q + (size_t)head * 65536;
  const float* Uh = U + (size_t)head * 65536;
  float s = 0.f;
  for (int i = 0; i < 256; ++i) s += (float)Qh[i * 256 + k] * Uh[i * 256 + k];
  ninv[head * 256 + k] = 1.0f / fmaxf(s, 1e-6f);
}

// Qp = Q * ninv[k];  WQ = Qp * U   (both f16)
__global__ void k_qp_wq(const _Float16* __restrict__ Q, const float* __restrict__ U,
                        const float* __restrict__ ninv, _Float16* __restrict__ Qp,
                        _Float16* __restrict__ WQ) {
  int i = blockIdx.x * 256 + threadIdx.x;  // 0..262143
  int k = i & 255, head = i >> 16;
  float q = (float)Q[i] * ninv[head * 256 + k];
  Qp[i] = (_Float16)q;
  WQ[i] = (_Float16)(q * U[i]);
}

// W = Qp @ R^T; WP = P .* W   (f16 out)
__global__ void k_W(const _Float16* __restrict__ Qp, const _Float16* __restrict__ R,
                    const _Float16* __restrict__ P, _Float16* __restrict__ WP) {
  int lane = threadIdx.x & 31;
  int wave = blockIdx.x * 4 + (threadIdx.x >> 5);  // 0..1023
  int head = wave >> 8, tile = wave & 255;
  int tm = tile >> 4, tn = tile & 15;
  const _Float16* Qh = Qp + (size_t)head * 65536;
  const _Float16* Rh = R + (size_t)head * 65536;
  const _Float16* Ph = P + (size_t)head * 65536;
  v8f acc = {};
#pragma unroll
  for (int kk = 0; kk < 8; ++kk) {
    if (kk < 7) {
      __builtin_prefetch(Qh + (size_t)(tm * 16 + (lane & 15)) * 256 + (kk + 1) * 32, 0, 1);
      __builtin_prefetch(Rh + (size_t)(tn * 16 + (lane & 15)) * 256 + (kk + 1) * 32, 0, 1);
    }
    v16h A = load_a16(Qh, 256, tm * 16, kk * 32, lane);
    v16h B = load_bt16(Rh, 256, tn * 16, kk * 32, lane);
    acc = WMMA_F16(A, B, acc);
  }
  _Float16* Wh = WP + (size_t)head * 65536;
  int nl = lane & 15, half = lane >> 4;
#pragma unroll
  for (int vv = 0; vv < 8; ++vv) {
    int m = tm * 16 + vv + 8 * half, n = tn * 16 + nl;
    Wh[m * 256 + n] = (_Float16)((float)Ph[m * 256 + n] * acc[vv]);
  }
}

// x1 = WP @ vT^T ; x2 = WQ @ vT^T   (f32 out, X12[sel][h][256][32])
__global__ void k_xv(const _Float16* __restrict__ WP, const _Float16* __restrict__ WQ,
                     const _Float16* __restrict__ abcv, float* __restrict__ X12) {
  int lane = threadIdx.x & 31;
  int wave = blockIdx.x * 4 + (threadIdx.x >> 5);  // 0..255
  int tn = wave & 1;
  int tm = (wave >> 1) & 15;
  int head = (wave >> 5) & 3;
  int sel = wave >> 7;
  const _Float16* Wm = (sel ? WQ : WP) + (size_t)head * 65536;
  const _Float16* vT = abcv + (size_t)(12 + head) * 8192;  // [32][256]
  v8f acc = {};
#pragma unroll
  for (int kk = 0; kk < 8; ++kk) {
    v16h A = load_a16(Wm, 256, tm * 16, kk * 32, lane);
    v16h B = load_bt16(vT, 256, tn * 16, kk * 32, lane);
    acc = WMMA_F16(A, B, acc);
  }
  float* dst = X12 + (size_t)(sel * 4 + head) * 8192;
  int nl = lane & 15, half = lane >> 4;
#pragma unroll
  for (int vv = 0; vv < 8; ++vv)
    dst[(tm * 16 + vv + 8 * half) * 32 + tn * 16 + nl] = acc[vv];
}

// y[n][h*32+d] = x1*x2  (f16)
__global__ void k_ymul(const float* __restrict__ X12, _Float16* __restrict__ Yh) {
  int i = blockIdx.x * 256 + threadIdx.x;  // 0..32767
  int n = i >> 7, c = i & 127;
  int head = c >> 5, d = c & 31;
  float x1 = X12[(size_t)(0 + head) * 8192 + n * 32 + d];
  float x2 = X12[(size_t)(4 + head) * 8192 + n * 32 + d];
  Yh[i] = (_Float16)(x1 * x2);
}

// out = Yh(256x128) @ WpT(128x128)^T + bias   (f32 out)
__global__ void k_proj(const _Float16* __restrict__ Yh, const _Float16* __restrict__ wpT,
                       const float* __restrict__ bias, float* __restrict__ out) {
  int lane = threadIdx.x & 31;
  int wave = blockIdx.x * 4 + (threadIdx.x >> 5);  // 0..127
  int tm = wave >> 3, tn = wave & 7;
  v8f acc = {};
#pragma unroll
  for (int kk = 0; kk < 4; ++kk) {
    v16h A = load_a16(Yh, 128, tm * 16, kk * 32, lane);
    v16h B = load_bt16(wpT, 128, tn * 16, kk * 32, lane);
    acc = WMMA_F16(A, B, acc);
  }
  int nl = lane & 15, half = lane >> 4;
#pragma unroll
  for (int vv = 0; vv < 8; ++vv) {
    int m = tm * 16 + vv + 8 * half, n = tn * 16 + nl;
    out[m * 128 + n] = acc[vv] + bias[n];
  }
}

// ---------------------------------------------------------------------------
// Launch
// ---------------------------------------------------------------------------
extern "C" void kernel_launch(void* const* d_in, const int* in_sizes, int n_in,
                              void* d_out, int out_size, void* d_ws, size_t ws_size,
                              hipStream_t stream) {
  const float* x  = (const float*)d_in[0];   // 256x128
  const float* Ww = (const float*)d_in[1];   // 128x512 (in, out)
  const float* Wp = (const float*)d_in[2];   // 128x128 (in, out)
  const float* bp = (const float*)d_in[3];   // 128

  char* ws = (char*)d_ws;
  _Float16* xh   = (_Float16*)(ws + 0);         // 64 KB
  _Float16* WhT  = (_Float16*)(ws + 65536);     // 128 KB (512x128)
  _Float16* WpT  = (_Float16*)(ws + 196608);    // 32 KB  (128x128)
  _Float16* abcv = (_Float16*)(ws + 229376);    // 256 KB: a,b,c [h][256][32]; vT [h][32][256]
  _Float16* P    = (_Float16*)(ws + 491520);    // 512 KB
  _Float16* Q    = (_Float16*)(ws + 1015808);   // 512 KB
  _Float16* R    = (_Float16*)(ws + 1540096);   // 512 KB
  _Float16* Rt   = (_Float16*)(ws + 2064384);   // 512 KB
  float*    U    = (float*)   (ws + 2588672);   // 1 MB
  float*    ninv = (float*)   (ws + 3637248);   // 4 KB
  _Float16* Qp   = (_Float16*)(ws + 3641344);   // 512 KB
  _Float16* WQ   = (_Float16*)(ws + 4165632);   // 512 KB
  _Float16* WP   = (_Float16*)(ws + 4689920);   // 512 KB
  float*    X12  = (float*)   (ws + 5214208);   // 256 KB
  _Float16* Yh   = (_Float16*)(ws + 5476352);   // 64 KB -> total 5541888 B (~5.3 MB)

  k_f32_to_f16<<<128, 256, 0, stream>>>(x, xh, 32768);
  k_f32_to_f16_T<<<256, 256, 0, stream>>>(Ww, WhT, 128, 512);
  k_f32_to_f16_T<<<64, 256, 0, stream>>>(Wp, WpT, 128, 128);
  k_qkv<<<128, 128, 0, stream>>>(xh, WhT, abcv);
  k_scores<<<1024, 128, 0, stream>>>(abcv, P, Q, R, Rt);
  k_U<<<256, 128, 0, stream>>>(P, Rt, U);
  k_norminv<<<4, 256, 0, stream>>>(Q, U, ninv);
  k_qp_wq<<<1024, 256, 0, stream>>>(Q, U, ninv, Qp, WQ);
  k_W<<<256, 128, 0, stream>>>(Qp, R, P, WP);
  k_xv<<<64, 128, 0, stream>>>(WP, WQ, abcv, X12);
  k_ymul<<<128, 256, 0, stream>>>(X12, Yh);
  k_proj<<<32, 128, 0, stream>>>(Yh, WpT, bp, (float*)d_out);
}
